// CrossAttention_2061584302183
// MI455X (gfx1250) — matrix operand order repaired and checked by reading, not compiled, not run on hardware
//
#include <hip/hip_runtime.h>
#include <hip/hip_bf16.h>
#include <stdint.h>

// MI455X / gfx1250 cross-attention with spatial reduction.
// HBM-bound on the 536MB distance_mask stream (~23us @ 23.3 TB/s).
// - all GEMMs: bf16 v_wmma_f32_16x16x32_bf16, fp32 accumulate, ping-pong k-loop
// - mask streamed via global_load_async_to_lds_b128 (ASYNCcnt) double-buffered in LDS

typedef __attribute__((ext_vector_type(16))) __bf16 v16bf;
typedef __attribute__((ext_vector_type(8)))  __bf16 v8bf;
typedef __attribute__((ext_vector_type(8)))  float  v8f;

#define HEADS 8
#define DIMC  256
#define HD    32
#define LQ    4096
#define LS    4096

__device__ __forceinline__ v8f zero8() {
  v8f z;
#pragma unroll
  for (int i = 0; i < 8; ++i) z[i] = 0.f;
  return z;
}

// A operand, 16x32 bf16 tile. base points at (row0, k0) of a [rows][ld] bf16 array.
// lane = M row; K = 16*half + e (one contiguous 16-element run).
__device__ __forceinline__ v16bf load_mat_a(const __bf16* base, int ld, int lane) {
  const int row = lane & 15, hh = lane >> 4;
  const __bf16* p = base + (size_t)row * ld + hh * 16;
  v8bf lo = *(const v8bf*)p;
  v8bf hi = *(const v8bf*)(p + 8);
  v16bf r;
#pragma unroll
  for (int i = 0; i < 8; ++i) { r[i] = lo[i]; r[i + 8] = hi[i]; }
  return r;
}

// B operand, 32x16 bf16 tile from [N][K]-major storage. base points at (col0, k0).
// lane = N column; K = 16*half + e (one contiguous 16-element run).
__device__ __forceinline__ v16bf load_mat_b(const __bf16* base, int ld, int lane) {
  const int colx = lane & 15, hh = lane >> 4;
  const __bf16* p = base + (size_t)colx * ld + hh * 16;
  v8bf lo = *(const v8bf*)p;
  v8bf hi = *(const v8bf*)(p + 8);
  v16bf r;
#pragma unroll
  for (int i = 0; i < 8; ++i) { r[i] = lo[i]; r[i + 8] = hi[i]; }
  return r;
}

__device__ __forceinline__ v8f wmma_bf16(v16bf a, v16bf b, v8f c) {
  return __builtin_amdgcn_wmma_f32_16x16x32_bf16(false, a, false, b, (short)0, c,
                                                 false, false);
}

// ---------------- prep kernels ----------------

__global__ void cvt_bf16(const float* __restrict__ in, __bf16* __restrict__ out, int n) {
  int i = blockIdx.x * blockDim.x + threadIdx.x;
  if (i < n) out[i] = (__bf16)in[i];
}

// conv_w[co][ci][kh][kw] -> Wc[co][t], t = (kh*2+kw)*256 + ci  (B operand [N][K])
__global__ void pack_convw(const float* __restrict__ w, __bf16* __restrict__ out) {
  int i = blockIdx.x * blockDim.x + threadIdx.x;           // 256*1024
  int co = i >> 10, t = i & 1023;
  int tap = t >> 8, ci = t & 255;
  out[i] = (__bf16)w[((co * 256 + ci) * 2 + (tap >> 1)) * 2 + (tap & 1)];
}

// im2col patches: Xp[ls][t] = x[(2*oh+kh)*128 + (2*ow+kw)][ci]
__global__ void pack_patches(const float* __restrict__ x, __bf16* __restrict__ out) {
  int i = blockIdx.x * blockDim.x + threadIdx.x;           // 4096*1024
  int ls = i >> 10, t = i & 1023;
  int tap = t >> 8, ci = t & 255;
  int oh = ls >> 6, ow = ls & 63;
  int ih = oh * 2 + (tap >> 1), iw = ow * 2 + (tap & 1);
  out[i] = (__bf16)x[(ih * 128 + iw) * 256 + ci];
}

// ---------------- generic WMMA GEMM: C = A[M][K] * B[N][K]^T ----------------
// K must be a multiple of 64. Ping-pong (unroll-2) k-loop: buffer set 0 computes
// while set 1 loads and vice versa -> no rotation register moves.
// mode 0: bf16 out [M][N], + bias
// mode 1: bf16 out [h][n][32]  (q layout), * scale
// mode 2: bf16 out [h][n][32]  (k layout)
// mode 3: bf16 out [h][32][n]  (v transposed)
// mode 4: f32  out [M][N], + bias
__global__ void gemm_wmma(const __bf16* __restrict__ A, const __bf16* __restrict__ B,
                          const float* __restrict__ bias, void* __restrict__ outp,
                          int M, int N, int K, int mode, float scale)
{
  const int lane = threadIdx.x & 31;
  const int wave = threadIdx.x >> 5;
  const int ntile = N >> 6;                     // 64-wide N super-tiles
  const int wt = blockIdx.x * 8 + wave;
  const int m0 = (wt / ntile) << 4;
  const int nb = (wt % ntile) << 6;
  if (m0 >= M) return;
  const int cl = lane & 15, hh = lane >> 4;

  const __bf16* Arow = A + (size_t)m0 * K;

  v8f acc[4];
#pragma unroll
  for (int j = 0; j < 4; ++j) acc[j] = zero8();

  v16bf a0 = load_mat_a(Arow, K, lane);
  v16bf b0[4];
#pragma unroll
  for (int j = 0; j < 4; ++j)
    b0[j] = load_mat_b(B + (size_t)(nb + 16 * j) * K, K, lane);

  for (int kb = 0; kb < K; kb += 64) {
    // load set 1 (kb+32) while set 0 computes
    v16bf a1 = load_mat_a(Arow + kb + 32, K, lane);
    v16bf b1[4];
#pragma unroll
    for (int j = 0; j < 4; ++j)
      b1[j] = load_mat_b(B + (size_t)(nb + 16 * j) * K + kb + 32, K, lane);
#pragma unroll
    for (int j = 0; j < 4; ++j) acc[j] = wmma_bf16(a0, b0[j], acc[j]);

    // load set 0 (kb+64) while set 1 computes
    if (kb + 64 < K) {
      a0 = load_mat_a(Arow + kb + 64, K, lane);
#pragma unroll
      for (int j = 0; j < 4; ++j)
        b0[j] = load_mat_b(B + (size_t)(nb + 16 * j) * K + kb + 64, K, lane);
    }
#pragma unroll
    for (int j = 0; j < 4; ++j) acc[j] = wmma_bf16(a1, b1[j], acc[j]);
  }

  __bf16* ob = (__bf16*)outp;
  float*  of = (float*)outp;
#pragma unroll
  for (int j = 0; j < 4; ++j) {
#pragma unroll
    for (int r = 0; r < 8; ++r) {
      int ri = m0 + r + 8 * hh;
      int ci = nb + 16 * j + cl;
      float v = acc[j][r];
      if (mode == 0) {
        ob[(size_t)ri * N + ci] = (__bf16)(v + bias[ci]);
      } else if (mode == 1 || mode == 2) {
        int hd = ci >> 5, d = ci & 31;
        ob[((size_t)hd * M + ri) * 32 + d] = (__bf16)(v * scale);
      } else if (mode == 3) {
        int hd = ci >> 5, d = ci & 31;
        ob[((size_t)hd * 32 + d) * M + ri] = (__bf16)v;
      } else {
        of[(size_t)ri * N + ci] = v + bias[ci];
      }
    }
  }
}

// ---------------- flash attention ----------------
// Q: [h][q][32] bf16 (scale pre-folded), K: [h][key][32] bf16, V: [h][32][key] bf16,
// mask: [h][q][key] f32 streamed via async global->LDS copies, O: [q][256] bf16.
// One wave = one 16-query tile; 32-key steps; 4 WMMAs per step.
__global__ void flash_attn(const __bf16* __restrict__ Q, const __bf16* __restrict__ Km,
                           const __bf16* __restrict__ Vm, const float* __restrict__ mask,
                           __bf16* __restrict__ O)
{
  __shared__ float  mstage[8][2][16 * 32];      // per-wave mask double buffer (4KB)
  __shared__ __bf16 pstage[8][16 * 32];         // per-wave P staging (1KB)
  const int lane = threadIdx.x & 31, wave = threadIdx.x >> 5;
  const int tile = blockIdx.x * 8 + wave;       // 2048 tiles = 8 heads * 256 q-tiles
  const int h  = tile >> 8;
  const int q0 = (tile & 255) << 4;
  const int cl = lane & 15, hh = lane >> 4;

  const __bf16* qbase = Q  + (size_t)h * LQ * HD;
  const __bf16* kbase = Km + (size_t)h * LS * HD;
  const __bf16* vbase = Vm + (size_t)h * HD * LS;
  const float*  mrow  = mask + ((size_t)h * LQ + q0) * LS;

  // async copy of one 16x32 f32 mask tile (2KB): 4 x 512B, one b128 per lane
  const uint32_t mlds0 = (uint32_t)(uintptr_t)&mstage[wave][0][0];
  const int olane = lane * 16;                  // byte slot within each 512B part
  auto issue_mask = [&](int kb2, int buf) {
#pragma unroll
    for (int part = 0; part < 4; ++part) {
      const int o = part * 512 + olane;         // byte offset within 2KB tile
      const int row = o >> 7;                   // 128B per mask row
      uint32_t la = mlds0 + (uint32_t)buf * 2048u + (uint32_t)o;
      uint64_t ga = (uint64_t)(uintptr_t)(mrow + (size_t)row * LS + kb2 + ((o & 127) >> 2));
      asm volatile("global_load_async_to_lds_b128 %0, %1, off"
                   :: "v"(la), "v"(ga) : "memory");
    }
  };

  v16bf qa = load_mat_a(qbase + (size_t)q0 * HD, HD, lane);   // loop-invariant
  v8f o0 = zero8(), o1 = zero8();
  float m[8], l[8];
#pragma unroll
  for (int r = 0; r < 8; ++r) { m[r] = -3.0e38f; l[r] = 0.f; }

  __bf16* pl = pstage[wave];
  issue_mask(0, 0);                             // prime the pipeline

  for (int kb = 0; kb < LS; kb += 32) {
    const int buf = (kb >> 5) & 1;
    if (kb + 32 < LS) {
      issue_mask(kb + 32, buf ^ 1);             // stream next tile while computing
      asm volatile("s_wait_asynccnt 0x4" ::: "memory");   // current tile landed
    } else {
      asm volatile("s_wait_asynccnt 0x0" ::: "memory");
    }

    // hoisted operand loads (overlap with mask wait / wmma)
    v16bf bk0 = load_mat_b(kbase + (size_t)kb * HD, HD, lane);
    v16bf bk1 = load_mat_b(kbase + (size_t)(kb + 16) * HD, HD, lane);
    v16bf bv0 = load_mat_b(vbase + kb, LS, lane);
    v16bf bv1 = load_mat_b(vbase + (size_t)16 * LS + kb, LS, lane);

    // S = Q * K^T  (scale already folded into Q)
    v8f s0 = wmma_bf16(qa, bk0, zero8());
    v8f s1 = wmma_bf16(qa, bk1, zero8());

    // + distance_mask from LDS (C layout: row = r+8*half, col = lane&15)
    const float* mt = &mstage[wave][buf][0];
#pragma unroll
    for (int r = 0; r < 8; ++r) {
      s0[r] += mt[(r + 8 * hh) * 32 + cl];
      s1[r] += mt[(r + 8 * hh) * 32 + 16 + cl];
    }

    // online softmax: row reductions across the 16 lanes of each half
#pragma unroll
    for (int r = 0; r < 8; ++r) {
      float mx = fmaxf(s0[r], s1[r]);
      mx = fmaxf(mx, __shfl_xor(mx, 1, 32));
      mx = fmaxf(mx, __shfl_xor(mx, 2, 32));
      mx = fmaxf(mx, __shfl_xor(mx, 4, 32));
      mx = fmaxf(mx, __shfl_xor(mx, 8, 32));
      float mn = fmaxf(m[r], mx);
      float al = __expf(m[r] - mn);
      m[r] = mn;
      float p0 = __expf(s0[r] - mn), p1 = __expf(s1[r] - mn);
      s0[r] = p0; s1[r] = p1;
      float rs = p0 + p1;
      rs += __shfl_xor(rs, 1, 32);
      rs += __shfl_xor(rs, 2, 32);
      rs += __shfl_xor(rs, 4, 32);
      rs += __shfl_xor(rs, 8, 32);
      l[r] = l[r] * al + rs;
      o0[r] *= al; o1[r] *= al;
    }

    // C-layout P -> LDS -> A-layout P (per-wave private staging)
#pragma unroll
    for (int r = 0; r < 8; ++r) {
      pl[(r + 8 * hh) * 32 + cl]      = (__bf16)s0[r];
      pl[(r + 8 * hh) * 32 + 16 + cl] = (__bf16)s1[r];
    }
    asm volatile("s_wait_dscnt 0x0" ::: "memory");
    v16bf pa = load_mat_a(pl, 32, lane);

    // O += P * V   (V stored [d][key] -> direct B-operand loads)
    o0 = wmma_bf16(pa, bv0, o0);
    o1 = wmma_bf16(pa, bv1, o1);
  }

  // normalize + store into [q][h*32+d] bf16 for the projection GEMM
#pragma unroll
  for (int r = 0; r < 8; ++r) {
    float inv = 1.f / l[r];
    int qr = q0 + r + 8 * hh;
    O[(size_t)qr * DIMC + h * HD + cl]      = (__bf16)(o0[r] * inv);
    O[(size_t)qr * DIMC + h * HD + 16 + cl] = (__bf16)(o1[r] * inv);
  }
}

// ---------------- launch ----------------

extern "C" void kernel_launch(void* const* d_in, const int* in_sizes, int n_in,
                              void* d_out, int out_size, void* d_ws, size_t ws_size,
                              hipStream_t stream)
{
  const float* x     = (const float*)d_in[0];   // [16384][256]
  const float* y     = (const float*)d_in[1];   // [4096][256]
  const float* mask  = (const float*)d_in[2];   // [8][4096][4096]
  const float* Wq    = (const float*)d_in[3];   // [256][256]
  const float* Wk    = (const float*)d_in[4];
  const float* Wv    = (const float*)d_in[5];
  const float* Wp    = (const float*)d_in[6];
  const float* bproj = (const float*)d_in[7];
  const float* cw    = (const float*)d_in[8];   // [256][256][2][2]
  const float* cb    = (const float*)d_in[9];

  char* w = (char*)d_ws;
  size_t off = 0;
  auto alloc = [&](size_t bytes) {
    void* p = w + off;
    off = (off + bytes + 255) & ~(size_t)255;
    return p;
  };
  __bf16* y_b  = (__bf16*)alloc((size_t)4096 * 256 * 2);
  __bf16* Wq_b = (__bf16*)alloc((size_t)256 * 256 * 2);
  __bf16* Wk_b = (__bf16*)alloc((size_t)256 * 256 * 2);
  __bf16* Wv_b = (__bf16*)alloc((size_t)256 * 256 * 2);
  __bf16* Wp_b = (__bf16*)alloc((size_t)256 * 256 * 2);
  __bf16* Wc_b = (__bf16*)alloc((size_t)256 * 1024 * 2);
  __bf16* Xp   = (__bf16*)alloc((size_t)4096 * 1024 * 2);
  __bf16* xs_b = (__bf16*)alloc((size_t)4096 * 256 * 2);
  __bf16* q_hb = (__bf16*)alloc((size_t)4096 * 256 * 2);
  __bf16* k_hb = (__bf16*)alloc((size_t)4096 * 256 * 2);
  __bf16* v_hb = (__bf16*)alloc((size_t)4096 * 256 * 2);
  __bf16* o_b  = (__bf16*)alloc((size_t)4096 * 256 * 2);

  const float scale = 0.17677669529663687f;     // 32^-0.5

  cvt_bf16<<<4096, 256, 0, stream>>>(y, y_b, 4096 * 256);
  cvt_bf16<<<256, 256, 0, stream>>>(Wq, Wq_b, 256 * 256);
  cvt_bf16<<<256, 256, 0, stream>>>(Wk, Wk_b, 256 * 256);
  cvt_bf16<<<256, 256, 0, stream>>>(Wv, Wv_b, 256 * 256);
  cvt_bf16<<<256, 256, 0, stream>>>(Wp, Wp_b, 256 * 256);
  pack_convw<<<1024, 256, 0, stream>>>(cw, Wc_b);
  pack_patches<<<16384, 256, 0, stream>>>(x, Xp);

  // spatial-reduction conv as GEMM: xs = Xp[4096][1024] * Wc^T + cb  -> bf16
  gemm_wmma<<<128, 256, 0, stream>>>(Xp, Wc_b, cb, xs_b, 4096, 256, 1024, 0, 1.f);
  // q = (y*Wq^T)*scale -> [h][n][32];  k -> [h][n][32];  v -> [h][32][n]
  gemm_wmma<<<128, 256, 0, stream>>>(y_b,  Wq_b, nullptr, q_hb, 4096, 256, 256, 1, scale);
  gemm_wmma<<<128, 256, 0, stream>>>(xs_b, Wk_b, nullptr, k_hb, 4096, 256, 256, 2, 1.f);
  gemm_wmma<<<128, 256, 0, stream>>>(xs_b, Wv_b, nullptr, v_hb, 4096, 256, 256, 3, 1.f);

  flash_attn<<<256, 256, 0, stream>>>(q_hb, k_hb, v_hb, mask, o_b);

  // out = o * Wproj^T + bproj  -> f32 d_out
  gemm_wmma<<<128, 256, 0, stream>>>(o_b, Wp_b, bproj, d_out, 4096, 256, 256, 4, 1.f);
}
